// RangeAttention_86414741995527
// MI455X (gfx1250) — matrix operand order repaired and checked by reading, not compiled, not run on hardware
//
#include <hip/hip_runtime.h>

#ifndef __has_builtin
#define __has_builtin(x) 0
#endif

#define ALPHA_SLOPE 0.2f
#define BN_EPS 1e-5f

typedef __attribute__((ext_vector_type(16))) __bf16        v16bf;
typedef __attribute__((ext_vector_type(8)))  float         v8f;
typedef __attribute__((ext_vector_type(4)))  unsigned int  u32x4;
typedef __attribute__((ext_vector_type(4)))  int           v4i;

#if defined(__HIP_DEVICE_COMPILE__)
typedef __attribute__((address_space(1))) v4i GV4;   // global v4i
typedef __attribute__((address_space(3))) v4i LV4;   // LDS v4i
#endif

union Frag32B { u32x4 q[2]; v16bf f; };

static constexpr int BB = 2, HH = 64, WW = 2048, CC = 64;
static constexpr int BH = BB * HH;        // 128
static constexpr int WP = WW + 2;         // 2050 : conv input, w-pad 1 each side
static constexpr int WF = WW + 6;         // 2054 : attention, w-pad 3 each side

static constexpr int WPK_ELEMS = 9 * 2 * 4 * 32 * 16;   // 36864 bf16 = 73728 B

// workspace layout (bytes)
static constexpr size_t XP_OFF  = 0;                              // bf16 [BH][WP][CC]
static constexpr size_t XP_SZ   = (size_t)BH * WP * CC * 2;
static constexpr size_t F_OFF   = XP_OFF + XP_SZ;                 // bf16 [BH][WF][CC]
static constexpr size_t F_SZ    = (size_t)BH * WF * CC * 2;
static constexpr size_t P_OFF   = F_OFF + F_SZ;                   // f32  [BH][WF][4]
static constexpr size_t P_SZ    = (size_t)BH * WF * 4 * 4;
static constexpr size_t WPK_OFF = P_OFF + P_SZ;                   // bf16 packed B-frags: [9][2][4][32 lanes][16]
static constexpr size_t WPK_SZ  = (size_t)WPK_ELEMS * 2;
static constexpr size_t S_OFF   = WPK_OFF + WPK_SZ;               // f32 [64] scale
static constexpr size_t T_OFF   = S_OFF + 256;                    // f32 [64] shift

__device__ __forceinline__ unsigned short f2bf(float x) {
    unsigned int u = __float_as_uint(x);
    unsigned int r = u + 0x7FFFu + ((u >> 16) & 1u);   // round-to-nearest-even
    return (unsigned short)(r >> 16);
}
__device__ __forceinline__ float bf2f_lo(unsigned int v) { return __uint_as_float(v << 16); }
__device__ __forceinline__ float bf2f_hi(unsigned int v) { return __uint_as_float(v & 0xFFFF0000u); }

// ---- prep: feats NCHW f32 -> padded NHWC bf16 ----------------------------
__global__ void k_prep_input(const float* __restrict__ feats, unsigned short* __restrict__ xp) {
    int wp = blockIdx.x * 64 + threadIdx.x;
    int bh = blockIdx.y;
    if (wp >= WP) return;
    unsigned short* dst = xp + ((size_t)bh * WP + wp) * CC;   // 128B aligned
    if (wp == 0 || wp == WP - 1) {
        u32x4 z = {0u, 0u, 0u, 0u};
#pragma unroll
        for (int c = 0; c < CC; c += 8) *(u32x4*)(dst + c) = z;
        return;
    }
    int b = bh >> 6, h = bh & 63, w = wp - 1;
    const float* src = feats + (((size_t)b * CC) * HH + h) * WW + w;   // + ci*HH*WW
#pragma unroll
    for (int ci = 0; ci < CC; ci += 8) {
        u32x4 q;
#pragma unroll
        for (int j = 0; j < 4; ++j) {
            unsigned int lo = f2bf(src[(size_t)(ci + 2 * j + 0) * HH * WW]);
            unsigned int hi = f2bf(src[(size_t)(ci + 2 * j + 1) * HH * WW]);
            q[j] = lo | (hi << 16);
        }
        *(u32x4*)(dst + ci) = q;
    }
}

// ---- prep: padded pos = concat(range, point_cloud) -----------------------
__global__ void k_prep_pos(const float* __restrict__ ri, const float* __restrict__ pc,
                           float* __restrict__ P) {
    int wp = blockIdx.x * 64 + threadIdx.x;
    int bh = blockIdx.y;
    if (wp >= WF) return;
    float* dst = P + ((size_t)bh * WF + wp) * 4;
    if (wp < 3 || wp >= WF - 3) { dst[0] = dst[1] = dst[2] = dst[3] = 0.f; return; }
    int w = wp - 3;
    size_t i = (size_t)bh * WW + w;
    dst[0] = ri[i]; dst[1] = pc[i * 3 + 0]; dst[2] = pc[i * 3 + 1]; dst[3] = pc[i * 3 + 2];
}

// ---- prep: pack conv weights into per-lane WMMA B fragments --------------
// B is K(32) x N(16) bf16: lane n=lane%16 holds K = 8*(lane/16)+j (j<8), 16+8*(lane/16)+(j-8) (j>=8) (32B blob).
__global__ void k_prep_w(const float* __restrict__ cw, unsigned short* __restrict__ wpk) {
    int tap   = blockIdx.x;           // kh*3+kw
    int tid   = threadIdx.x;          // 256 = 2*4*32
    int chunk = tid >> 7;
    int ct    = (tid >> 5) & 3;
    int lane  = tid & 31;
    int c     = ct * 16 + (lane & 15);
    int kbase = chunk * 32 + (lane >> 4) * 8;
    unsigned short* dst = wpk + (((size_t)(tap * 2 + chunk) * 4 + ct) * 32 + lane) * 16;
    for (int j = 0; j < 16; ++j) {
        int ci = kbase + j + (j >= 8 ? 8 : 0);
        dst[j] = f2bf(cw[((size_t)c * CC + ci) * 9 + tap]);   // W[c][ci][kh][kw]
    }
}

// ---- prep: fold conv bias + BN into per-channel scale/shift --------------
__global__ void k_prep_bn(const float* __restrict__ cb, const float* __restrict__ g,
                          const float* __restrict__ be, const float* __restrict__ mn,
                          const float* __restrict__ vr, float* __restrict__ s,
                          float* __restrict__ t) {
    int c = threadIdx.x;
    float inv = g[c] * rsqrtf(vr[c] + BN_EPS);
    s[c] = inv;
    t[c] = (cb[c] - mn[c]) * inv + be[c];
}

// ---- prep: zero f's w-halo (3 each side) ---------------------------------
__global__ void k_zero_fpad(unsigned short* __restrict__ F) {
    int bh = blockIdx.x;
    int t = threadIdx.x;              // 384 = 6*64
    int p = t >> 6, c = t & 63;
    int wf = (p < 3) ? p : (WF - 6 + p);
    F[((size_t)bh * WF + wf) * CC + c] = 0;
}

// ---- conv 3x3 + BN as bf16 WMMA GEMM -------------------------------------
// wave: 16 w-positions (M) x 64 channels (4 N-tiles), K = 9 taps * 64 ci.
// Packed weight fragments are staged once per workgroup into LDS via async
// global->LDS DMA and consumed via ds_load_b128.
__global__ void __launch_bounds__(128)
k_conv_wmma(const unsigned short* __restrict__ xp,
            const unsigned short* __restrict__ wpk,
            const float* __restrict__ s, const float* __restrict__ t,
            unsigned short* __restrict__ F) {
    __shared__ __align__(16) unsigned short wlds[WPK_ELEMS];   // 73728 B

    const int lane = threadIdx.x & 31;
    const int wave = threadIdx.x >> 5;        // 4 waves / block
    const int bh = blockIdx.y;
    const int h  = bh & (HH - 1);
    const int w0 = (blockIdx.x * 4 + wave) * 16;
    const int m  = lane & 15;                 // A-row (w offset)
    const int hi = lane >> 4;

    // ---- stage packed weights into LDS (72 KB, 36 b128 per thread) ----
    {
        const char* src = (const char*)wpk;
        char*       dst = (char*)wlds;
#if defined(__HIP_DEVICE_COMPILE__) && __has_builtin(__builtin_amdgcn_global_load_async_to_lds_b128)
        for (int i = threadIdx.x; i < WPK_ELEMS / 8; i += 128) {
            __builtin_amdgcn_global_load_async_to_lds_b128(
                (GV4*)(src + (size_t)i * 16),
                (LV4*)(dst + (size_t)i * 16),
                0, 0);
        }
#if __has_builtin(__builtin_amdgcn_s_wait_asynccnt)
        __builtin_amdgcn_s_wait_asynccnt(0);
#else
        asm volatile("s_wait_asynccnt 0x0" ::: "memory");
#endif
#else
        for (int i = threadIdx.x; i < WPK_ELEMS / 8; i += 128)
            *(u32x4*)(dst + (size_t)i * 16) = *(const u32x4*)(src + (size_t)i * 16);
#endif
        __syncthreads();
    }

    v8f acc[4] = {};

#pragma unroll
    for (int kh = 0; kh < 3; ++kh) {
        int hsrc = h + kh - 1;
        if (hsrc >= 0 && hsrc < HH) {         // wave-uniform: EXEC stays all-1s
            int bhs = (bh & ~(HH - 1)) | hsrc;
#pragma unroll
            for (int kw = 0; kw < 3; ++kw) {
                int tap = kh * 3 + kw;
#pragma unroll
                for (int chunk = 0; chunk < 2; ++chunk) {
                    // A fragment: 16-bit A 16x32 layout; per lane two b128 loads
                    const char* rowbase = (const char*)xp +
                        (((size_t)bhs * WP + (w0 + m + kw)) * CC + chunk * 32) * 2 + hi * 16;
                    Frag32B A;
                    A.q[0] = *(const u32x4*)(rowbase);
                    A.q[1] = *(const u32x4*)(rowbase + 32);
                    const char* wb = (const char*)wlds +
                        ((size_t)(tap * 2 + chunk) * 4) * 1024 + (size_t)lane * 32;
#pragma unroll
                    for (int ct = 0; ct < 4; ++ct) {
                        Frag32B Bf;
                        const u32x4* bp = (const u32x4*)(wb + (size_t)ct * 1024);
                        Bf.q[0] = bp[0]; Bf.q[1] = bp[1];
                        acc[ct] = __builtin_amdgcn_wmma_f32_16x16x32_bf16(
                            false, A.f, false, Bf.f, (short)0, acc[ct], false, false);
                    }
                }
            }
        }
    }

    // epilogue: BN + store padded bf16 f.  C/D layout: VGPR r, lane L ->
    // M = r + 8*(L/16), N = L%16.
#pragma unroll
    for (int ct = 0; ct < 4; ++ct) {
        int c = ct * 16 + m;
        float sc = s[c], tb = t[c];
#pragma unroll
        for (int r = 0; r < 8; ++r) {
            int w = w0 + r + 8 * hi;
            float v = acc[ct][r] * sc + tb;
            F[((size_t)bh * WF + (w + 3)) * CC + c] = f2bf(v);
        }
    }
}

// ---- attention: 6-neighbor softmax over per-channel scores ---------------
// thread = 1 pixel x 8 channels
__global__ void k_attn(const float* __restrict__ P, const unsigned short* __restrict__ F,
                       const float* __restrict__ amat, float* __restrict__ out) {
    int tid = threadIdx.x;                    // 256
    int bh = blockIdx.y;
    int b = bh >> 6, h = bh & 63;
    int w  = blockIdx.x * 32 + (tid >> 3);
    int cg = (tid & 7) * 8;
    size_t rowP = (size_t)bh * WF;

    const float* ps = P + (rowP + w + 3) * 4;
    float p0 = ps[0], p1 = ps[1], p2 = ps[2], p3 = ps[3];

    float A0[8], A1[8], A2[8], A3[8];
#pragma unroll
    for (int j = 0; j < 8; ++j) {
        A0[j] = amat[0 * CC + cg + j];
        A1[j] = amat[1 * CC + cg + j];
        A2[j] = amat[2 * CC + cg + j];
        A3[j] = amat[3 * CC + cg + j];
    }

    float e[6][8], mx[8];
#pragma unroll
    for (int j = 0; j < 8; ++j) mx[j] = -3.4e38f;
#pragma unroll
    for (int n = 0; n < 6; ++n) {             // neighbor offsets n-2 in {-2..3}
        const float* pn = P + (rowP + w + 1 + n) * 4;
        float d0 = p0 - pn[0], d1 = p1 - pn[1], d2 = p2 - pn[2], d3 = p3 - pn[3];
#pragma unroll
        for (int j = 0; j < 8; ++j) {
            float v = d0 * A0[j] + d1 * A1[j] + d2 * A2[j] + d3 * A3[j];
            v = v > 0.f ? v : ALPHA_SLOPE * v;
            e[n][j] = v;
            mx[j] = fmaxf(mx[j], v);
        }
    }
    float den[8];
#pragma unroll
    for (int j = 0; j < 8; ++j) den[j] = 0.f;
#pragma unroll
    for (int n = 0; n < 6; ++n)
#pragma unroll
        for (int j = 0; j < 8; ++j) {
            float ex = __expf(e[n][j] - mx[j]);
            e[n][j] = ex;
            den[j] += ex;
        }
    float acc[8];
#pragma unroll
    for (int j = 0; j < 8; ++j) acc[j] = 0.f;
#pragma unroll
    for (int n = 0; n < 6; ++n) {
        // 8 contiguous bf16 = one b128 (16B-aligned: row*64 + cg, cg % 8 == 0)
        u32x4 q = *(const u32x4*)(F + (rowP + w + 1 + n) * CC + cg);
#pragma unroll
        for (int j2 = 0; j2 < 4; ++j2) {
            acc[2 * j2 + 0] += e[n][2 * j2 + 0] * bf2f_lo(q[j2]);
            acc[2 * j2 + 1] += e[n][2 * j2 + 1] * bf2f_hi(q[j2]);
        }
    }
#pragma unroll
    for (int j = 0; j < 8; ++j)
        out[(((size_t)b * CC + cg + j) * HH + h) * WW + w] = acc[j] / den[j];
}

extern "C" void kernel_launch(void* const* d_in, const int* in_sizes, int n_in,
                              void* d_out, int out_size, void* d_ws, size_t ws_size,
                              hipStream_t stream) {
    (void)in_sizes; (void)n_in; (void)out_size; (void)ws_size;
    const float* range = (const float*)d_in[0];
    const float* pcld  = (const float*)d_in[1];
    const float* feats = (const float*)d_in[2];
    const float* convw = (const float*)d_in[3];
    const float* convb = (const float*)d_in[4];
    const float* gamma = (const float*)d_in[5];
    const float* beta  = (const float*)d_in[6];
    const float* mean  = (const float*)d_in[7];
    const float* var   = (const float*)d_in[8];
    const float* amat  = (const float*)d_in[9];
    // d_in[10] = kernel_size (7): compile-time specialized (6 neighbors).

    char* ws = (char*)d_ws;
    unsigned short* xp  = (unsigned short*)(ws + XP_OFF);
    unsigned short* Fb  = (unsigned short*)(ws + F_OFF);
    float*          P   = (float*)(ws + P_OFF);
    unsigned short* wpk = (unsigned short*)(ws + WPK_OFF);
    float*          s   = (float*)(ws + S_OFF);
    float*          t   = (float*)(ws + T_OFF);
    float*          out = (float*)d_out;

    k_prep_input<<<dim3((WP + 63) / 64, BH), 64, 0, stream>>>(feats, xp);
    k_prep_pos  <<<dim3((WF + 63) / 64, BH), 64, 0, stream>>>(range, pcld, P);
    k_prep_w    <<<dim3(9), 256, 0, stream>>>(convw, wpk);
    k_prep_bn   <<<dim3(1), 64, 0, stream>>>(convb, gamma, beta, mean, var, s, t);
    k_zero_fpad <<<dim3(BH), 384, 0, stream>>>(Fb);
    k_conv_wmma <<<dim3(WW / 64, BH), 128, 0, stream>>>(xp, wpk, s, t, Fb);
    k_attn      <<<dim3(WW / 32, BH), 256, 0, stream>>>(P, Fb, amat, out);
}
